// attention_gear_77506979824106
// MI455X (gfx1250) — matrix operand order repaired and checked by reading, not compiled, not run on hardware
//
#include <hip/hip_runtime.h>
#include <hip/hip_bf16.h>

// ---------------------------------------------------------------------------
// Flash-attention forward, causal. B=2, S=2048, H=32, D=128, fp32 in/out.
// bf16 WMMA (v_wmma_f32_16x16x32_bf16) for QK^T and PV, f32 accumulation.
// - 256-thread blocks (8 waves x 16 q-rows) share double-buffered LDS tiles
// - split load/convert staging = software pipeline over HBM latency
// - row-max reduction via ds_swizzle (xor), row-sum via a "ones" WMMA column
// - f32->bf16 pair pack via single v_perm_b32 (truncation; cvt_pk_bf16
//   builtin probe-confirmed absent on gfx1250)
// ---------------------------------------------------------------------------

typedef __attribute__((ext_vector_type(16))) __bf16 v16bf;
typedef __attribute__((ext_vector_type(8)))  float  v8f;

union U32x16 {            // one 16x32-bf16 WMMA fragment (8 VGPRs)
    unsigned int u32[8];
    uint4        u4[2];
    v16bf        bf;
};

// single-op f32 -> bf16 (truncate): take high 16 bits
static __device__ __forceinline__ unsigned short bf16t(float f) {
    return (unsigned short)(__float_as_uint(f) >> 16);
}

// single v_perm_b32: {hi[31:16], lo[31:16]} -> packed bf16 pair
static __device__ __forceinline__ unsigned int pk(float lo, float hi) {
    return __builtin_amdgcn_perm(__float_as_uint(hi), __float_as_uint(lo),
                                 0x07060302u);
}

// xor-swizzle across the 16-lane half-group: ds_swizzle group-of-32,
// new_lane = (lane & 0x1F) ^ XORM  (single DS op, immediate pattern)
template<int XORM>
static __device__ __forceinline__ float swz_xor(float x) {
    return __int_as_float(
        __builtin_amdgcn_ds_swizzle(__float_as_int(x), 0x1F | (XORM << 10)));
}

constexpr int Bc = 2, Sc = 2048, Hc = 32, Dc = 128;
constexpr int RS = Hc * Dc;          // element stride between sequence positions

// Online-softmax row update for one 16x32 score tile (two C-fragments).
// Computes new row max, alpha, exp(p), stores bf16 P to LDS. Row sums are NOT
// reduced here: the caller accumulates l = P . 1 with a WMMA.
template<bool MASK>
static __device__ __forceinline__ void softmax_update(
    v8f& s0, v8f& s1, float mi[8], float al[8],
    int j, int q0, int laneN, int laneHi, unsigned short* Pw)
{
#pragma unroll
    for (int r = 0; r < 8; ++r) {
        float x0 = s0[r], x1 = s1[r];
        if (MASK) {
            const int rowg = q0 + laneHi * 8 + r;
            x0 = (j + laneN      <= rowg) ? x0 : -1e9f;
            x1 = (j + laneN + 16 <= rowg) ? x1 : -1e9f;
        }
        float t = fmaxf(x0, x1);
        t = fmaxf(t, swz_xor<1>(t));
        t = fmaxf(t, swz_xor<2>(t));
        t = fmaxf(t, swz_xor<4>(t));
        t = fmaxf(t, swz_xor<8>(t));
        const float mnew = fmaxf(mi[r], t);
        al[r] = __expf(mi[r] - mnew);
        mi[r] = mnew;
        const float p0 = __expf(x0 - mnew);   // masked lanes underflow to 0
        const float p1 = __expf(x1 - mnew);
        // C-layout -> LDS (transpose staging for the GEMM2 A-fragment)
        Pw[(laneHi * 8 + r) * 40 + laneN]      = bf16t(p0);
        Pw[(laneHi * 8 + r) * 40 + laneN + 16] = bf16t(p1);
    }
}

__global__ void __launch_bounds__(256)
fa_fwd_causal_bf16wmma(const float* __restrict__ Q,
                       const float* __restrict__ K,
                       const float* __restrict__ V,
                       float* __restrict__ O)
{
    // Double-buffered tiles; padded strides (all 16B-aligned) to spread banks.
    __shared__ __align__(16) unsigned short Kls[2][32][136];   // K tile row-major bf16
    __shared__ __align__(16) unsigned short Vtls[2][128][40];  // V tile transposed [d][k]
    __shared__ __align__(16) unsigned short Pls[8][16][40];    // per-wave P scratch

    const int tid    = threadIdx.x;
    const int wave   = tid >> 5;
    const int lane   = tid & 31;
    const int laneN  = lane & 15;
    const int laneHi = lane >> 4;

    const int b   = blockIdx.z;
    const int h   = blockIdx.y;
    const int q0b = blockIdx.x * 128;      // 128 query rows per block (8 waves)
    const int q0  = q0b + wave * 16;

    const size_t base = (size_t)b * Sc * RS + (size_t)h * Dc;
    const float* Qb = Q + base;
    const float* Kb = K + base;
    const float* Vb = V + base;

    const int jtiles = (q0b >> 5) + 4;     // causal: keys 0 .. q0b+127

    // ---- Q fragments (A-layout), softmax scale folded in -------------------
    const float qscale = 0.08838834764831845f;   // 1/sqrt(128)
    v16bf aq[4];
    {
        const float* qrow = Qb + (size_t)(q0 + laneN) * RS;
#pragma unroll
        for (int ks = 0; ks < 4; ++ks) {
            U32x16 u;
#pragma unroll
            for (int g = 0; g < 2; ++g) {
                const int d0 = ks * 32 + laneHi * 16 + g * 8;
                float4 x = *(const float4*)(qrow + d0);
                float4 y = *(const float4*)(qrow + d0 + 4);
                u.u32[g * 4 + 0] = pk(x.x * qscale, x.y * qscale);
                u.u32[g * 4 + 1] = pk(x.z * qscale, x.w * qscale);
                u.u32[g * 4 + 2] = pk(y.x * qscale, y.y * qscale);
                u.u32[g * 4 + 3] = pk(y.z * qscale, y.w * qscale);
            }
            aq[ks] = u.bf;
        }
    }

    // constant "ones" B-fragment: l = P . 1 (row sums) via WMMA, no LDS reads
    U32x16 ones;
#pragma unroll
    for (int i = 0; i < 8; ++i) ones.u32[i] = 0x3F803F80u;   // bf16 1.0 pair

    // ---- staging registers (K: 4 x float4, V: 16 floats) --------------------
    const int sr  = tid >> 3;               // K staging row 0..31
    const int scn = (tid & 7) * 16;         // K staging column chunk (16 elems)
    const int vn  = tid & 127;              // V staging column d
    const int vkh = (tid >> 7) * 16;        // V staging k half (16 elems)
    float4 sk[4];
    float  svf[16];

    auto stage_load = [&](int jt) {
        const int j = jt * 32;
        const float* krow = Kb + (size_t)(j + sr) * RS + scn;
#pragma unroll
        for (int t = 0; t < 4; ++t) sk[t] = *(const float4*)(krow + t * 4);
        const float* vc = Vb + (size_t)(j + vkh) * RS + vn;
#pragma unroll
        for (int i = 0; i < 16; ++i) svf[i] = vc[i * RS];
        if (jt + 1 < jtiles) {                          // L2 prefetch of tile jt+1
            __builtin_prefetch(krow + 32 * RS, 0, 1);
            __builtin_prefetch(vc + 32 * RS, 0, 1);
        }
    };

    auto stage_store = [&](int buf) {
        {
            uint4 u0, u1;
            u0.x = pk(sk[0].x, sk[0].y); u0.y = pk(sk[0].z, sk[0].w);
            u0.z = pk(sk[1].x, sk[1].y); u0.w = pk(sk[1].z, sk[1].w);
            u1.x = pk(sk[2].x, sk[2].y); u1.y = pk(sk[2].z, sk[2].w);
            u1.z = pk(sk[3].x, sk[3].y); u1.w = pk(sk[3].z, sk[3].w);
            *(uint4*)&Kls[buf][sr][scn]     = u0;
            *(uint4*)&Kls[buf][sr][scn + 8] = u1;
        }
        {
            uint4 u0, u1;
            u0.x = pk(svf[0],  svf[1]);  u0.y = pk(svf[2],  svf[3]);
            u0.z = pk(svf[4],  svf[5]);  u0.w = pk(svf[6],  svf[7]);
            u1.x = pk(svf[8],  svf[9]);  u1.y = pk(svf[10], svf[11]);
            u1.z = pk(svf[12], svf[13]); u1.w = pk(svf[14], svf[15]);
            *(uint4*)&Vtls[buf][vn][vkh]     = u0;
            *(uint4*)&Vtls[buf][vn][vkh + 8] = u1;
        }
    };

    // ---- running state ------------------------------------------------------
    v8f accv[8], lv;                       // O accumulators + row-sum accumulator
    float mi[8];
#pragma unroll
    for (int nt = 0; nt < 8; ++nt)
#pragma unroll
        for (int r = 0; r < 8; ++r) accv[nt][r] = 0.0f;
#pragma unroll
    for (int r = 0; r < 8; ++r) { lv[r] = 0.0f; mi[r] = -1e30f; }

    unsigned short* Pw = &Pls[wave][0][0];

    // prologue: stage tile 0 into buffer 0
    stage_load(0);
    stage_store(0);

    for (int jt = 0; jt < jtiles; ++jt) {
        const int buf  = jt & 1;
        const int j    = jt * 32;
        const bool more = (jt + 1 < jtiles);

        __syncthreads();                    // tile jt fully staged for everyone
        if (more) stage_load(jt + 1);       // issue next tile's global loads now

        // ---- GEMM1: preload all 8 K fragments, then 8 WMMAs ----------------
        U32x16 kb[8];
#pragma unroll
        for (int ks = 0; ks < 4; ++ks) {
            const int d0 = ks * 32 + laneHi * 16;
            kb[2 * ks + 0].u4[0] = *(const uint4*)&Kls[buf][laneN][d0];
            kb[2 * ks + 0].u4[1] = *(const uint4*)&Kls[buf][laneN][d0 + 8];
            kb[2 * ks + 1].u4[0] = *(const uint4*)&Kls[buf][laneN + 16][d0];
            kb[2 * ks + 1].u4[1] = *(const uint4*)&Kls[buf][laneN + 16][d0 + 8];
        }
        v8f s0, s1;
#pragma unroll
        for (int r = 0; r < 8; ++r) { s0[r] = 0.0f; s1[r] = 0.0f; }
#pragma unroll
        for (int ks = 0; ks < 4; ++ks) {
            s0 = __builtin_amdgcn_wmma_f32_16x16x32_bf16(false, aq[ks], false, kb[2 * ks + 0].bf,
                                                         (short)0, s0, false, false);
            s1 = __builtin_amdgcn_wmma_f32_16x16x32_bf16(false, aq[ks], false, kb[2 * ks + 1].bf,
                                                         (short)0, s1, false, false);
        }

        // ---- online softmax (mask only on diagonal-straddling tiles) -------
        float al[8];
        if (j + 31 <= q0)
            softmax_update<false>(s0, s1, mi, al, j, q0, laneN, laneHi, Pw);
        else
            softmax_update<true >(s0, s1, mi, al, j, q0, laneN, laneHi, Pw);

        v8f alv;
#pragma unroll
        for (int r = 0; r < 8; ++r) alv[r] = al[r];
#pragma unroll
        for (int nt = 0; nt < 8; ++nt) accv[nt] *= alv;
        lv *= alv;

        // wave-local LDS ordering: P stores visible before re-reading
        asm volatile("s_wait_dscnt 0x0" ::: "memory");

        // ---- GEMM2: P fragment + 8 V fragments + ones fragment -------------
        U32x16 pfrag;
        pfrag.u4[0] = *(const uint4*)&Pw[laneN * 40 + laneHi * 16];
        pfrag.u4[1] = *(const uint4*)&Pw[laneN * 40 + laneHi * 16 + 8];

        U32x16 vf[8];
#pragma unroll
        for (int nt = 0; nt < 8; ++nt) {
            const unsigned short* vr = &Vtls[buf][nt * 16 + laneN][laneHi * 16];
            vf[nt].u4[0] = *(const uint4*)vr;
            vf[nt].u4[1] = *(const uint4*)(vr + 8);
        }
        // row sums: lv += P . 1 (constant B fragment, no LDS traffic)
        lv = __builtin_amdgcn_wmma_f32_16x16x32_bf16(false, pfrag.bf, false, ones.bf,
                                                     (short)0, lv, false, false);
#pragma unroll
        for (int nt = 0; nt < 8; ++nt) {
            accv[nt] = __builtin_amdgcn_wmma_f32_16x16x32_bf16(false, pfrag.bf, false, vf[nt].bf,
                                                               (short)0, accv[nt], false, false);
        }

        // ---- convert + store next tile into the other LDS buffer -----------
        if (more) stage_store(buf ^ 1);
    }

    // ---- epilogue: O = acc / l, back to [B,S,H,D] ---------------------------
#pragma unroll
    for (int r = 0; r < 8; ++r) {
        const float invl = 1.0f / lv[r];
        float* orow = O + (size_t)b * Sc * RS
                        + (size_t)(q0 + laneHi * 8 + r) * RS
                        + (size_t)h * Dc;
#pragma unroll
        for (int nt = 0; nt < 8; ++nt)
            orow[nt * 16 + laneN] = accv[nt][r] * invl;
    }
}

extern "C" void kernel_launch(void* const* d_in, const int* in_sizes, int n_in,
                              void* d_out, int out_size, void* d_ws, size_t ws_size,
                              hipStream_t stream) {
    (void)in_sizes; (void)n_in; (void)out_size; (void)d_ws; (void)ws_size;
    const float* q = (const float*)d_in[0];
    const float* k = (const float*)d_in[1];
    const float* v = (const float*)d_in[2];
    // d_in[3]=start_pos(0), d_in[4]=seqlen, d_in[5]=mask: causal handled in-kernel
    float* o = (float*)d_out;
    dim3 grid(Sc / 128, Hc, Bc);  // (q-blocks, heads, batch)
    dim3 block(256);              // 8 waves of 32
    fa_fwd_causal_bf16wmma<<<grid, block, 0, stream>>>(q, k, v, o);
}
